// Centripetal_mask_55765855371795
// MI455X (gfx1250) — matrix operand order repaired and checked by reading, not compiled, not run on hardware
//
#include <hip/hip_runtime.h>
#include <cstdint>

// ---------------------------------------------------------------------------
// Types
// ---------------------------------------------------------------------------
typedef __bf16 bf16;
typedef bf16  v16bf __attribute__((ext_vector_type(16)));
typedef float v8f   __attribute__((ext_vector_type(8)));
typedef unsigned int u32x4 __attribute__((ext_vector_type(4)));
typedef int   i32x8 __attribute__((ext_vector_type(8)));
typedef int   i32x4 __attribute__((ext_vector_type(4)));

#if __has_builtin(__builtin_amdgcn_tensor_load_to_lds) && __has_builtin(__builtin_amdgcn_s_wait_tensorcnt)
#define HAVE_TDM 1
#else
#define HAVE_TDM 0
#endif

// Epilogue modes
#define EPI_BN_RELU_BF16     0   // bf16 out = relu(g*acc + b)
#define EPI_BIAS_RELU_BF16   1   // bf16 out = relu(acc + b)
#define EPI_BN_ADD_RELU_BF16 2   // bf16 out = relu(g*acc + b + addsrc)
#define EPI_BN_F32           3   // f32  out = g*acc + b
#define EPI_BIAS_F32         4   // f32  out = acc + b
#define EPI_BF16             5   // bf16 out = acc

// ---------------------------------------------------------------------------
// WMMA fragment helpers (wave32, 16x16x32 bf16)
// A tile in LDS: [128 rows(cout-local)][32 k]  row-major
// B tile (conv): [64 n(spatial-local)][32 k]   row-major (transposed staging)
// B tile (gemm): [32 k][64 n]                  row-major (TDM-friendly)
// ---------------------------------------------------------------------------
__device__ __forceinline__ v16bf load_a_frag(const bf16* lA, int rowLocal, int lane) {
  v16bf a;
  const int kh = (lane >> 4) * 16;
  const bf16* p = lA + rowLocal * 32;
#pragma unroll
  for (int j = 0; j < 16; ++j) a[j] = p[kh + j];
  return a;
}

__device__ __forceinline__ v16bf load_b_frag_nk(const bf16* lB, int nLocal, int lane) {
  v16bf b;
  const int k0 = (lane >> 4) * 16;
  const bf16* p = lB + nLocal * 32 + k0;
#pragma unroll
  for (int j = 0; j < 16; ++j) b[j] = p[j];
  return b;
}

__device__ __forceinline__ v16bf load_b_frag_kn(const bf16* lB, int nLocal, int lane) {
  v16bf b;
  const int k0 = (lane >> 4) * 16;
#pragma unroll
  for (int j = 0; j < 16; ++j) b[j] = lB[(k0 + j) * 64 + nLocal];
  return b;
}

__device__ __forceinline__ void epilogue_store(
    float a, int co, int pos, int Cout, int HW, int epi,
    const float* g, const float* b, const float* addsrc, void* out)
{
  if (co >= Cout) return;
  switch (epi) {
    case EPI_BN_RELU_BF16: {
      float y = fmaxf(g[co] * a + b[co], 0.0f);
      ((bf16*)out)[(size_t)co * HW + pos] = (bf16)y;
    } break;
    case EPI_BIAS_RELU_BF16: {
      float y = fmaxf(a + b[co], 0.0f);
      ((bf16*)out)[(size_t)co * HW + pos] = (bf16)y;
    } break;
    case EPI_BN_ADD_RELU_BF16: {
      float y = fmaxf(g[co] * a + b[co] + addsrc[(size_t)co * HW + pos], 0.0f);
      ((bf16*)out)[(size_t)co * HW + pos] = (bf16)y;
    } break;
    case EPI_BN_F32: {
      ((float*)out)[(size_t)co * HW + pos] = g[co] * a + b[co];
    } break;
    case EPI_BIAS_F32: {
      ((float*)out)[(size_t)co * HW + pos] = a + b[co];
    } break;
    default: {
      ((bf16*)out)[(size_t)co * HW + pos] = (bf16)a;
    } break;
  }
}

// ---------------------------------------------------------------------------
// Implicit-GEMM convolution (ksize = 1 or 3, pad = ksize/2), bf16 WMMA.
// Block tile: 128 Cout x 64 spatial. 8 waves = 4 (cout) x 2 (spatial) groups;
// each wave does a 32x32 tile = 2x2 WMMA subtiles (4 wmma / K-step, frag reuse).
// act: bf16 [Cin][H*W]; w: fp32 [Cout][Cin*ks*ks] (converted to bf16 in LDS)
// ---------------------------------------------------------------------------
__global__ __launch_bounds__(256) void conv_wmma_kernel(
    const bf16* __restrict__ act, const float* __restrict__ w,
    const float* __restrict__ g, const float* __restrict__ b,
    const float* __restrict__ addsrc, void* __restrict__ out,
    int Cin, int Cout, int ksize, int H, int W, int epi)
{
  const int HW = H * W;
  const int K = Cin * ksize * ksize;

  __shared__ bf16 lA[128 * 32];   // 8 KB
  __shared__ bf16 lB[64 * 32];    // 4 KB

  const int t    = threadIdx.x;
  const int lane = t & 31;
  const int wid  = t >> 5;
  const int coutBase = blockIdx.y * 128;
  const int posBase  = blockIdx.x * 64;
  const int wCout = (wid & 3) * 32;   // wave's 32-row cout group
  const int wSp   = (wid >> 2) * 32;  // wave's 32-col spatial group

  v8f acc[4];
#pragma unroll
  for (int i = 0; i < 4; ++i) acc[i] = (v8f){0.f, 0.f, 0.f, 0.f, 0.f, 0.f, 0.f, 0.f};

  const int aRow = t >> 1;          // 0..127
  const int aK   = (t & 1) * 16;    // 0 or 16 (16 consecutive k per thread)
  const int bK   = t >> 3;          // 0..31
  const int bN   = (t & 7) * 8;     // 0..56

  for (int kk = 0; kk < K; kk += 32) {
    __syncthreads();
    // ---- stage A: weights fp32 -> bf16 (contiguous 16-wide per thread) ----
    {
      const int co = coutBase + aRow;
      const float* wp = w + (size_t)co * K + kk + aK;
      if (kk + 32 < K && co < Cout)
        __builtin_prefetch(wp + 32, 0, 1);   // global_prefetch_b8: next K-step
#pragma unroll
      for (int j = 0; j < 16; ++j) {
        float wv = (co < Cout) ? wp[j] : 0.0f;
        lA[aRow * 32 + aK + j] = (bf16)wv;
      }
    }
    // ---- stage B: im2col gather of bf16 activations ----
    {
      const int kg = kk + bK;
      int cin, dy, dx;
      if (ksize == 3) {
        cin = kg / 9;
        const int tap = kg - cin * 9;
        const int ty = tap / 3;
        dy = ty - 1;
        dx = (tap - ty * 3) - 1;
      } else {
        cin = kg; dy = 0; dx = 0;
      }
      const bf16* ap = act + (size_t)cin * HW;
#pragma unroll
      for (int j = 0; j < 8; ++j) {
        const int pos = posBase + bN + j;
        const int row = pos / W;
        const int col = pos - row * W;
        const int sr = row + dy, sc = col + dx;
        bf16 v = (bf16)0.0f;
        if (sr >= 0 && sr < H && sc >= 0 && sc < W) v = ap[sr * W + sc];
        lB[(bN + j) * 32 + bK] = v;
      }
    }
    __syncthreads();

    v16bf af[2], bfm[2];
    af[0]  = load_a_frag(lA, wCout + (lane & 15), lane);
    af[1]  = load_a_frag(lA, wCout + 16 + (lane & 15), lane);
    bfm[0] = load_b_frag_nk(lB, wSp + (lane & 15), lane);
    bfm[1] = load_b_frag_nk(lB, wSp + 16 + (lane & 15), lane);
#pragma unroll
    for (int ai = 0; ai < 2; ++ai)
#pragma unroll
      for (int bi = 0; bi < 2; ++bi)
        acc[ai * 2 + bi] = __builtin_amdgcn_wmma_f32_16x16x32_bf16(
            false, af[ai], false, bfm[bi], (short)0, acc[ai * 2 + bi], false, false);
  }

  // ---- epilogue ----
  const int mh = (lane >> 4) * 8;
  const int nx = lane & 15;
#pragma unroll
  for (int ai = 0; ai < 2; ++ai)
#pragma unroll
    for (int bi = 0; bi < 2; ++bi)
#pragma unroll
      for (int v = 0; v < 8; ++v) {
        const int co  = coutBase + wCout + ai * 16 + mh + v;
        const int pos = posBase + wSp + bi * 16 + nx;
        epilogue_store(acc[ai * 2 + bi][v], co, pos, Cout, HW, epi, g, b, addsrc, out);
      }
}

// ---------------------------------------------------------------------------
// Dense GEMM over a precomputed im2col panel (deformable conv einsum):
//   out[co][mBase+m] = sum_k w[co][k] * Bbuf[k][m],  K=2304, bf16 WMMA.
// Block tile: 128 Cout x 64 m, 2x2 WMMA subtiles per wave. B tiles
// (32 k x 64 m, contiguous rows) are pulled LDS-side by the Tensor Data
// Mover (D# built per ISA §8), else cooperative copy.
// ---------------------------------------------------------------------------
__global__ __launch_bounds__(256) void gemm_wmma_tdm_kernel(
    const bf16* __restrict__ Bbuf, const float* __restrict__ w,
    bf16* __restrict__ out, int K, int Cout, int Mc, int mBase, int HW)
{
  __shared__ bf16 lA[128 * 32];
  __shared__ bf16 lBt[32 * 64];

  const int t    = threadIdx.x;
  const int lane = t & 31;
  const int wid  = t >> 5;
  const int coutBase = blockIdx.y * 128;
  const int nBase    = blockIdx.x * 64;
  const int wCout = (wid & 3) * 32;
  const int wSp   = (wid >> 2) * 32;

  v8f acc[4];
#pragma unroll
  for (int i = 0; i < 4; ++i) acc[i] = (v8f){0.f, 0.f, 0.f, 0.f, 0.f, 0.f, 0.f, 0.f};

  const int aRow = t >> 1;
  const int aK   = (t & 1) * 16;

  for (int kk = 0; kk < K; kk += 32) {
    __syncthreads();
    // ---- stage A: dcn weights fp32 -> bf16 ----
    {
      const int co = coutBase + aRow;
      const float* wp = w + (size_t)co * K + kk + aK;
      if (kk + 32 < K && co < Cout)
        __builtin_prefetch(wp + 32, 0, 1);
#pragma unroll
      for (int j = 0; j < 16; ++j) {
        float wv = (co < Cout) ? wp[j] : 0.0f;
        lA[aRow * 32 + aK + j] = (bf16)wv;
      }
    }
    // ---- stage B: 32x64 bf16 tile via TDM (one DMA per block, wave 0) ----
#if HAVE_TDM
    if (wid == 0) {
      const uint64_t gaddr = (uint64_t)(uintptr_t)(Bbuf + (size_t)kk * Mc + nBase);
      const uint32_t laddr = (uint32_t)(uintptr_t)(&lBt[0]);
      u32x4 g0;
      g0[0] = 1u;                                   // count=1, user descriptor
      g0[1] = laddr;                                // lds_addr
      g0[2] = (uint32_t)gaddr;                      // global_addr[31:0]
      g0[3] = ((uint32_t)(gaddr >> 32) & 0x01FFFFFFu) | (2u << 30); // addr[56:32] | type=2
      i32x8 g1;
      const uint32_t td0 = (uint32_t)Mc;            // tensor_dim0 (elements)
      const uint32_t td1 = (uint32_t)K;             // tensor_dim1
      g1[0] = (int)(1u << 16);                      // data_size = 1 (2 bytes/elem)
      g1[1] = (int)((td0 & 0xFFFFu) << 16);         // tensor_dim0[15:0]
      g1[2] = (int)(((td0 >> 16) & 0xFFFFu) | ((td1 & 0xFFFFu) << 16));
      g1[3] = (int)(((td1 >> 16) & 0xFFFFu) | (64u << 16));  // tile_dim0 = 64
      g1[4] = (int)32u;                             // tile_dim1 = 32, tile_dim2 = 0
      g1[5] = (int)(uint32_t)Mc;                    // tensor_dim0_stride[31:0]
      g1[6] = 0;                                    // stride hi / dim1_stride
      g1[7] = 0;
      const i32x4 gz4 = {0, 0, 0, 0};               // 2D tensor: groups 2/3 unused
      const i32x8 gz8 = {0, 0, 0, 0, 0, 0, 0, 0};   // clang-23 lane: extra group arg
      __builtin_amdgcn_tensor_load_to_lds(g0, g1, gz4, gz4, gz8, 0);
      __builtin_amdgcn_s_wait_tensorcnt(0);
    }
#else
    {
      const int bK = t >> 3;
      const int bN = (t & 7) * 8;
#pragma unroll
      for (int j = 0; j < 8; ++j)
        lBt[bK * 64 + bN + j] = Bbuf[(size_t)(kk + bK) * Mc + nBase + bN + j];
    }
#endif
    __syncthreads();

    v16bf af[2], bfm[2];
    af[0]  = load_a_frag(lA, wCout + (lane & 15), lane);
    af[1]  = load_a_frag(lA, wCout + 16 + (lane & 15), lane);
    bfm[0] = load_b_frag_kn(lBt, wSp + (lane & 15), lane);
    bfm[1] = load_b_frag_kn(lBt, wSp + 16 + (lane & 15), lane);
#pragma unroll
    for (int ai = 0; ai < 2; ++ai)
#pragma unroll
      for (int bi = 0; bi < 2; ++bi)
        acc[ai * 2 + bi] = __builtin_amdgcn_wmma_f32_16x16x32_bf16(
            false, af[ai], false, bfm[bi], (short)0, acc[ai * 2 + bi], false, false);
  }

  const int mh = (lane >> 4) * 8;
  const int nx = lane & 15;
#pragma unroll
  for (int ai = 0; ai < 2; ++ai)
#pragma unroll
    for (int bi = 0; bi < 2; ++bi)
#pragma unroll
      for (int v = 0; v < 8; ++v) {
        const int co = coutBase + wCout + ai * 16 + mh + v;
        if (co < Cout)
          out[(size_t)co * HW + mBase + nBase + wSp + bi * 16 + nx] = (bf16)acc[ai * 2 + bi][v];
      }
}

// ---------------------------------------------------------------------------
// Small support kernels
// ---------------------------------------------------------------------------
__global__ void cvt_f32_bf16_kernel(const float* __restrict__ in, bf16* __restrict__ out, int n) {
  const int i = blockIdx.x * blockDim.x + threadIdx.x;
  if (i < n) out[i] = (bf16)in[i];
}

__global__ void add_bf16_kernel(const bf16* __restrict__ a, const bf16* __restrict__ b,
                                bf16* __restrict__ o, int n) {
  const int i = blockIdx.x * blockDim.x + threadIdx.x;
  if (i < n) o[i] = (bf16)((float)a[i] + (float)b[i]);
}

// cummax along H (axis=2): one thread per (c, col)
__global__ void cummax_h_kernel(bf16* x, int C, int H, int W, int rev) {
  const int i = blockIdx.x * blockDim.x + threadIdx.x;
  if (i >= C * W) return;
  const int c = i / W;
  const int col = i - c * W;
  bf16* p = x + (size_t)c * H * W + col;
  float m = -3.4e38f;
  if (rev) {
    for (int h = H - 1; h >= 0; --h) { m = fmaxf(m, (float)p[(size_t)h * W]); p[(size_t)h * W] = (bf16)m; }
  } else {
    for (int h = 0; h < H; ++h)      { m = fmaxf(m, (float)p[(size_t)h * W]); p[(size_t)h * W] = (bf16)m; }
  }
}

// cummax along W (axis=3): one thread per (c, row)
__global__ void cummax_w_kernel(bf16* x, int C, int H, int W, int rev) {
  const int i = blockIdx.x * blockDim.x + threadIdx.x;
  if (i >= C * H) return;
  const int c = i / H;
  const int row = i - c * H;
  bf16* p = x + (size_t)c * H * W + (size_t)row * W;
  float m = -3.4e38f;
  if (rev) {
    for (int cw = W - 1; cw >= 0; --cw) { m = fmaxf(m, (float)p[cw]); p[cw] = (bf16)m; }
  } else {
    for (int cw = 0; cw < W; ++cw)      { m = fmaxf(m, (float)p[cw]); p[cw] = (bf16)m; }
  }
}

// offmap[o][p] = off_w[o][0]*offc[0][p] + off_w[o][1]*offc[1][p], o in [0,18)
__global__ void offmap_kernel(const float* __restrict__ offc, const float* __restrict__ ow,
                              float* __restrict__ offmap, int HW) {
  const int p = blockIdx.x * blockDim.x + threadIdx.x;
  if (p >= HW) return;
  const float c0 = offc[p];
  const float c1 = offc[HW + p];
#pragma unroll
  for (int o = 0; o < 18; ++o)
    offmap[(size_t)o * HW + p] = ow[o * 2 + 0] * c0 + ow[o * 2 + 1] * c1;
}

__device__ __forceinline__ int iclamp(int v, int lo, int hi) {
  return v < lo ? lo : (v > hi ? hi : v);
}

// Bilinear deform sampling into im2col panel samp[(c*9+k)][m], one thread per (k, m)
__global__ void deform_sample_kernel(const bf16* __restrict__ pool, const float* __restrict__ offmap,
                                     bf16* __restrict__ samp, int C, int H, int W, int mBase, int Mc) {
  const int idx = blockIdx.x * blockDim.x + threadIdx.x;
  if (idx >= 9 * Mc) return;
  const int k = idx / Mc;
  const int m = idx - k * Mc;
  const int HW = H * W;
  const int pos = mBase + m;
  const int row = pos / W;
  const int col = pos - row * W;

  const float dy = offmap[(size_t)(2 * k) * HW + pos];
  const float dx = offmap[(size_t)(2 * k + 1) * HW + pos];
  const float py = (float)row + (float)(k / 3 - 1) + dy;
  const float px = (float)col + (float)(k % 3 - 1) + dx;
  const float y0f = floorf(py), x0f = floorf(px);
  const int y0 = (int)y0f, x0 = (int)x0f;
  const float wy = py - y0f, wx = px - x0f;
  const int y1 = y0 + 1, x1 = x0 + 1;

  const float v00 = (y0 >= 0 && y0 < H && x0 >= 0 && x0 < W) ? (1.f - wy) * (1.f - wx) : 0.f;
  const float v01 = (y0 >= 0 && y0 < H && x1 >= 0 && x1 < W) ? (1.f - wy) * wx         : 0.f;
  const float v10 = (y1 >= 0 && y1 < H && x0 >= 0 && x0 < W) ? wy * (1.f - wx)         : 0.f;
  const float v11 = (y1 >= 0 && y1 < H && x1 >= 0 && x1 < W) ? wy * wx                 : 0.f;
  const int y0c = iclamp(y0, 0, H - 1), y1c = iclamp(y1, 0, H - 1);
  const int x0c = iclamp(x0, 0, W - 1), x1c = iclamp(x1, 0, W - 1);
  const int i00 = y0c * W + x0c, i01 = y0c * W + x1c;
  const int i10 = y1c * W + x0c, i11 = y1c * W + x1c;

  for (int c = 0; c < C; ++c) {
    const bf16* bp = pool + (size_t)c * HW;
    const float s = v00 * (float)bp[i00] + v01 * (float)bp[i01] +
                    v10 * (float)bp[i10] + v11 * (float)bp[i11];
    samp[(size_t)(c * 9 + k) * Mc + m] = (bf16)s;
  }
}

// ---------------------------------------------------------------------------
// Host orchestration
// ---------------------------------------------------------------------------
extern "C" void kernel_launch(void* const* d_in, const int* in_sizes, int n_in,
                              void* d_out, int out_size, void* d_ws, size_t ws_size,
                              hipStream_t stream) {
  (void)in_sizes; (void)n_in; (void)out_size; (void)ws_size;
  const int C = 256, H = 128, W = 128, HW = H * W;
  const int MC = 2048;  // deform-conv spatial chunk

  const float* feat0 = (const float*)d_in[0];
  const float* feat1 = (const float*)d_in[1];

  // params flattened in JAX pytree (sorted-key) order after feat0/feat1
  enum {
    P_C1_B = 2, P_C1_G, P_C1_W, P_C2_B, P_C2_G, P_C2_W, P_DCN_W,
    P_H_B1, P_H_B2, P_H_W1, P_H_W2,
    P_O_B1, P_O_B2, P_O_W1, P_O_W2,
    P_OC2_B1, P_OC2_B2, P_OC2_W1, P_OC2_W2,
    P_OC_B1, P_OC_B2, P_OC_W1, P_OC_W2,
    P_OFF_W, P_P1_B, P_P1_G, P_P1_W, P_P2_B, P_P2_G, P_P2_W,
    P_PC_B, P_PC_G, P_PC_W
  };
  auto prm = [&](int idx, int i, size_t per) -> const float* {
    return (const float*)d_in[idx] + (size_t)i * per;
  };

  // --- workspace carve-up (~90 MB) ---
  char* wsp = (char*)d_ws;
  auto carve = [&](size_t bytes) -> char* {
    char* p = wsp; wsp += (bytes + 255) & ~(size_t)255; return p;
  };
  bf16*  f0b   = (bf16*)carve((size_t)C * HW * 2);
  bf16*  f1b   = (bf16*)carve((size_t)C * HW * 2);
  bf16*  p1b   = (bf16*)carve((size_t)128 * HW * 2);
  bf16*  p2b   = (bf16*)carve((size_t)128 * HW * 2);
  bf16*  sb    = (bf16*)carve((size_t)128 * HW * 2);
  float* scf   = (float*)carve((size_t)C * HW * 4);
  bf16*  rb    = (bf16*)carve((size_t)C * HW * 2);
  bf16*  poolb = (bf16*)carve((size_t)C * HW * 2);
  bf16*  tb    = (bf16*)carve((size_t)C * HW * 2);
  float* offc  = (float*)carve((size_t)2 * HW * 4);
  float* offm  = (float*)carve((size_t)18 * HW * 4);
  bf16*  sampb = (bf16*)carve((size_t)2304 * MC * 2);
  bf16*  xfb   = (bf16*)carve((size_t)C * HW * 2);

  auto conv = [&](const bf16* act, const float* w, const float* g, const float* b,
                  const float* add, void* out, int Cin, int Cout, int ks, int epi) {
    dim3 grid(HW / 64, (Cout + 127) / 128);
    conv_wmma_kernel<<<grid, 256, 0, stream>>>(act, w, g, b, add, out, Cin, Cout, ks, H, W, epi);
  };

  const int nEl = C * HW;
  cvt_f32_bf16_kernel<<<(nEl + 255) / 256, 256, 0, stream>>>(feat0, f0b, nEl);
  cvt_f32_bf16_kernel<<<(nEl + 255) / 256, 256, 0, stream>>>(feat1, f1b, nEl);

  for (int i = 0; i < 4; ++i) {
    const bf16* xb = (i < 2) ? f1b : f0b;
    const int rev = (i % 2 == 0) ? 1 : 0;
    float* ob = (float*)d_out + (size_t)i * 86 * HW;

    // corner pool
    conv(xb, prm(P_P1_W, i, 294912), prm(P_P1_G, i, 128), prm(P_P1_B, i, 128),
         nullptr, p1b, 256, 128, 3, EPI_BN_RELU_BF16);
    cummax_h_kernel<<<(128 * W + 255) / 256, 256, 0, stream>>>(p1b, 128, H, W, rev);
    conv(xb, prm(P_P2_W, i, 294912), prm(P_P2_G, i, 128), prm(P_P2_B, i, 128),
         nullptr, p2b, 256, 128, 3, EPI_BN_RELU_BF16);
    cummax_w_kernel<<<(128 * H + 255) / 256, 256, 0, stream>>>(p2b, 128, H, W, rev);
    add_bf16_kernel<<<(128 * HW + 255) / 256, 256, 0, stream>>>(p1b, p2b, sb, 128 * HW);
    conv(xb, prm(P_C1_W, i, 65536), prm(P_C1_G, i, 256), prm(P_C1_B, i, 256),
         nullptr, scf, 256, 256, 1, EPI_BN_F32);
    conv(sb, prm(P_PC_W, i, 294912), prm(P_PC_G, i, 256), prm(P_PC_B, i, 256),
         scf, rb, 128, 256, 3, EPI_BN_ADD_RELU_BF16);
    conv(rb, prm(P_C2_W, i, 589824), prm(P_C2_G, i, 256), prm(P_C2_B, i, 256),
         nullptr, poolb, 256, 256, 3, EPI_BN_RELU_BF16);

    // heat head -> d_out[ch 0..79]
    conv(poolb, prm(P_H_W1, i, 589824), nullptr, prm(P_H_B1, i, 256), nullptr,
         tb, 256, 256, 3, EPI_BIAS_RELU_BF16);
    conv(tb, prm(P_H_W2, i, 20480), nullptr, prm(P_H_B2, i, 80), nullptr,
         ob, 256, 80, 1, EPI_BIAS_F32);

    // off_c head -> d_out[ch 80..81] (and ws copy for offmap)
    conv(poolb, prm(P_OC_W1, i, 589824), nullptr, prm(P_OC_B1, i, 256), nullptr,
         tb, 256, 256, 3, EPI_BIAS_RELU_BF16);
    conv(tb, prm(P_OC_W2, i, 512), nullptr, prm(P_OC_B2, i, 2), nullptr,
         ob + (size_t)80 * HW, 256, 2, 1, EPI_BIAS_F32);
    conv(tb, prm(P_OC_W2, i, 512), nullptr, prm(P_OC_B2, i, 2), nullptr,
         offc, 256, 2, 1, EPI_BIAS_F32);

    // offset map + deformable conv (chunked im2col GEMM, TDM-staged B panels)
    offmap_kernel<<<(HW + 255) / 256, 256, 0, stream>>>(offc, prm(P_OFF_W, i, 36), offm, HW);
    for (int chnk = 0; chnk < HW / MC; ++chnk) {
      const int mBase = chnk * MC;
      deform_sample_kernel<<<(9 * MC + 255) / 256, 256, 0, stream>>>(
          poolb, offm, sampb, C, H, W, mBase, MC);
      dim3 gg(MC / 64, (256 + 127) / 128);
      gemm_wmma_tdm_kernel<<<gg, 256, 0, stream>>>(
          sampb, prm(P_DCN_W, i, 589824), xfb, 2304, 256, MC, mBase, HW);
    }

    // off_c_2 head (on x_fadp) -> d_out[ch 82..83]
    conv(xfb, prm(P_OC2_W1, i, 589824), nullptr, prm(P_OC2_B1, i, 256), nullptr,
         tb, 256, 256, 3, EPI_BIAS_RELU_BF16);
    conv(tb, prm(P_OC2_W2, i, 512), nullptr, prm(P_OC2_B2, i, 2), nullptr,
         ob + (size_t)82 * HW, 256, 2, 1, EPI_BIAS_F32);

    // off head -> d_out[ch 84..85]
    conv(poolb, prm(P_O_W1, i, 589824), nullptr, prm(P_O_B1, i, 256), nullptr,
         tb, 256, 256, 3, EPI_BIAS_RELU_BF16);
    conv(tb, prm(P_O_W2, i, 512), nullptr, prm(P_O_B2, i, 2), nullptr,
         ob + (size_t)84 * HW, 256, 2, 1, EPI_BIAS_F32);
  }
}
